// ContrastiveLoss_31945966747953
// MI455X (gfx1250) — matrix operand order repaired and checked by reading, not compiled, not run on hardware
//
#include <hip/hip_runtime.h>
#include <hip/hip_bf16.h>
#include <math.h>

// ---------------------------------------------------------------------------
// CDNA5 / gfx1250 implementation of the contrastive point-sampling loss.
//   B=8, H=W=256 (HW=65536), Cf=128, n_cls derived from sizes (=2).
//   N_ANC=128, N_POS=256, N_NEG=1536 -> 1920 points per (b, class).
//   K padded 130 -> 160 (5 x 32 for v_wmma_f32_16x16x32_f16).
// ---------------------------------------------------------------------------

typedef __attribute__((ext_vector_type(16))) _Float16 v16h;
typedef __attribute__((ext_vector_type(8)))  float    v8f;

union Frag { int4 i4[2]; v16h h; };

#define SORT_N   32768
#define SORT_NT  1024
#define KPAD     160
#define NPTS     1920   // 128 anc + 256 pos + 1536 neg
#define NA       128
#define NT_POS   16     // 256 pos rows  = 16 N-tiles
#define NT_ALL   112    // + 1536 neg rows = 96 N-tiles

// ---------------------------------------------------------------------------
// Kernel 1: per (b, class, polarity) stable compaction of certainty values,
// then full bitonic sort of (value, compacted-index) pairs in LDS (256KB —
// uses CDNA5's 320KB-per-WGP LDS), then emit top-k index lists.
// ---------------------------------------------------------------------------
__global__ __launch_bounds__(SORT_NT)
void k_sort_select(const float* __restrict__ coarse, const int* __restrict__ GT,
                   int* __restrict__ selIdx, int nc, int HW) {
  extern __shared__ char smem[];
  float* sv    = (float*)smem;                          // SORT_N values
  int*   si    = (int*)(smem + (size_t)SORT_N * 4);     // SORT_N indices
  int*   scanb = (int*)(smem + (size_t)SORT_N * 8);     // SORT_NT scan temp

  const int tid = threadIdx.x;
  const int z2  = blockIdx.x;
  const int b   = z2 / (nc * 2);
  const int rr  = z2 % (nc * 2);
  const int cls = rr >> 1;
  const int isneg = rr & 1;
  const int z   = b * nc + cls;

  const float* cb = coarse + (size_t)b * nc * HW;
  const int*   gb = GT + (size_t)b * HW;

  // ---- stable row-major compaction (matches vals[mask] ordering) ----
  int base = 0;
  for (int j0 = 0; j0 < HW; j0 += SORT_NT) {
    int j = j0 + tid;
    int flag = 0; float cert = 0.0f;
    if (j < HW) {
      bool msk  = (gb[j] == cls);
      bool take = isneg ? (!msk) : msk;
      if (take) {
        flag = 1;
        float m1 = -3.402823466e38f, m2 = -3.402823466e38f;
        for (int c = 0; c < nc; ++c) {
          float vv = cb[(size_t)c * HW + j];
          if (vv > m1) { m2 = m1; m1 = vv; } else if (vv > m2) { m2 = vv; }
        }
        cert = m1 - m2;     // top1 - top2 logit margin
      }
    }
    scanb[tid] = flag;
    __syncthreads();
    for (int off = 1; off < SORT_NT; off <<= 1) {
      int a  = scanb[tid];
      int bs = (tid >= off) ? scanb[tid - off] : 0;
      __syncthreads();
      scanb[tid] = a + bs;
      __syncthreads();
    }
    int incl = scanb[tid];
    int tot  = scanb[SORT_NT - 1];
    if (flag) {
      int p = base + incl - 1;
      if (p < SORT_N) { sv[p] = cert; si[p] = p; }   // idx into COMPACTED array
    }
    base += tot;
    __syncthreads();
  }
  int cnt = base > SORT_N ? SORT_N : base;

  // pad (descending sort -> sentinel goes last)
  for (int p = cnt + tid; p < SORT_N; p += SORT_NT) {
    sv[p] = -3.402823466e38f; si[p] = 0x7FFFFFFF;
  }
  __syncthreads();

  // ---- bitonic sort, key = (-value, index) ascending == value descending ----
  for (int k2 = 2; k2 <= SORT_N; k2 <<= 1) {
    for (int jj = k2 >> 1; jj > 0; jj >>= 1) {
      for (int i = tid; i < SORT_N; i += SORT_NT) {
        int ixj = i ^ jj;
        if (ixj > i) {
          float va = sv[i], vb = sv[ixj];
          int   ia = si[i], ib = si[ixj];
          bool up = ((i & k2) == 0);
          bool iGreater = (va < vb) || (va == vb && ia > ib);  // key[i] > key[ixj]
          bool dosw = up ? iGreater : !iGreater;
          if (dosw) { sv[i] = vb; sv[ixj] = va; si[i] = ib; si[ixj] = ia; }
        }
      }
      __syncthreads();
    }
  }

  // ---- emit selected indices: [anc(128) | pos(256) | neg(1536)] per z ----
  int* outz = selIdx + (size_t)z * NPTS;
  if (!isneg) {
    for (int t = tid; t < 128; t += SORT_NT) outz[t] = si[t];          // top certainty
    int b0 = cnt - 256; if (b0 < 0) b0 = 0;
    for (int t = tid; t < 256; t += SORT_NT) outz[128 + t] = si[b0 + t]; // bottom certainty
  } else {
    for (int t = tid; t < 1536; t += SORT_NT) outz[384 + t] = si[t];   // top certainty (~mask)
  }
}

// ---------------------------------------------------------------------------
// Kernel 2: gather 130-ch feature at flat pixel = selected (compacted) index,
// L2-normalize across channels (eps=1e-8), store f16 rows padded to K=160.
// One wave32 per point; channel sum-of-squares via shfl_xor butterfly.
// ---------------------------------------------------------------------------
__global__ __launch_bounds__(128)
void k_gather_norm(const float* __restrict__ coarse, const float* __restrict__ fine,
                   const int* __restrict__ selIdx, _Float16* __restrict__ F,
                   int nc, int Cf, int HW, int totalPts) {
  int wave = blockIdx.x * (blockDim.x >> 5) + (threadIdx.x >> 5);
  int lane = threadIdx.x & 31;
  if (wave >= totalPts) return;
  int z = wave / NPTS;
  int s = wave - z * NPTS;
  int b = z / nc;
  int idx = selIdx[(size_t)z * NPTS + s];
  idx = idx < 0 ? 0 : (idx >= HW ? HW - 1 : idx);
  int CT = nc + Cf;

  float v[5]; float ss = 0.0f;
#pragma unroll
  for (int s2 = 0; s2 < 5; ++s2) {
    int c = lane + 32 * s2;
    float x = 0.0f;
    if (c < nc)      x = coarse[((size_t)b * nc + c) * HW + idx];
    else if (c < CT) x = fine[((size_t)b * Cf + (c - nc)) * HW + idx];
    v[s2] = x; ss += x * x;
  }
  ss += __shfl_xor(ss, 16, 32);
  ss += __shfl_xor(ss, 8, 32);
  ss += __shfl_xor(ss, 4, 32);
  ss += __shfl_xor(ss, 2, 32);
  ss += __shfl_xor(ss, 1, 32);
  float sc = 1.0f / fmaxf(sqrtf(ss), 1e-8f);

  _Float16* row = F + (size_t)((size_t)z * NPTS + s) * KPAD;
#pragma unroll
  for (int s2 = 0; s2 < 5; ++s2) {
    int c = lane + 32 * s2;
    if (c < KPAD) row[c] = (_Float16)(v[s2] * sc);
  }
}

// ---------------------------------------------------------------------------
// Kernel 3: per (b, class): sims = A(128x160) x B'(160x1792) via
// v_wmma_f32_16x16x32_f16.  All 5 B fragments are loaded into distinct
// registers before the wmma chain (loads clause + progressive waits overlap
// with wmma).  Pos (tiles 0..15) and neg (tiles 16..111) run as two
// straight-line loops -> no per-iteration predicate.  exp(sim) accumulates
// into per-lane registers; one butterfly + LDS store at the end.
// Wave w owns M-tile w -> single-writer, fully deterministic (no atomics).
// ---------------------------------------------------------------------------
__global__ __launch_bounds__(256)
void k_gemm_loss(const _Float16* __restrict__ F, float* __restrict__ partial) {
  __shared__ __align__(16) _Float16 Ash[NA * KPAD];   // 40960 B
  __shared__ float sums[256];                          // [0..127]=pos, [128..255]=neg
  __shared__ float red[128];

  const int tid = threadIdx.x;
  const int z   = blockIdx.x;
  const _Float16* Fz = F + (size_t)z * NPTS * KPAD;

  // stage anchor matrix A into LDS
  {
    const unsigned int* src = (const unsigned int*)Fz;
    unsigned int* dst = (unsigned int*)Ash;
    for (int t = tid; t < NA * KPAD / 2; t += 256) dst[t] = src[t];
  }
  __syncthreads();

  const int lane = tid & 31;
  const int n16  = lane & 15;
  const int hf   = lane >> 4;
  const int mt   = tid >> 5;      // wave id == M-tile it owns

  // A fragments (held for all 112 N-tiles).  A layout (ISA 7.12.2, 16-bit A
  // 16x32): lane M = lane&15; halves 0..7 -> K = hf*8+{0..7}, 8..15 -> +16.
  v16h afr[5];
#pragma unroll
  for (int kc = 0; kc < 5; ++kc) {
    const _Float16* ap = &Ash[(mt * 16 + n16) * KPAD + kc * 32 + hf * 8];
    Frag u;
    u.i4[0] = *(const int4*)(ap);
    u.i4[1] = *(const int4*)(ap + 16);
    afr[kc] = u.h;
  }

  // per-lane exp-sum accumulators: r -> M = mt*16 + hf*8 + r, this lane's N
  float accP[8], accN[8];
#pragma unroll
  for (int r2 = 0; r2 < 8; ++r2) { accP[r2] = 0.0f; accN[r2] = 0.0f; }

  // ---- positive tiles (rows 128..383) ----
  for (int nt = 0; nt < NT_POS; ++nt) {
    const _Float16* bp0 = Fz + (size_t)(NA + nt * 16 + n16) * KPAD;
    __builtin_prefetch(Fz + (size_t)(NA + (nt + 1) * 16 + n16) * KPAD, 0, 3);

    Frag u[5];
#pragma unroll
    for (int kc = 0; kc < 5; ++kc) {       // B layout: lane N = lane&15; K = hf*8+{0..7}, +16
      const _Float16* bp = bp0 + kc * 32 + hf * 8;
      u[kc].i4[0] = *(const int4*)(bp);
      u[kc].i4[1] = *(const int4*)(bp + 16);
    }
    v8f acc = {0.f, 0.f, 0.f, 0.f, 0.f, 0.f, 0.f, 0.f};
#pragma unroll
    for (int kc = 0; kc < 5; ++kc)
      acc = __builtin_amdgcn_wmma_f32_16x16x32_f16(
          false, afr[kc], false, u[kc].h, (short)0, acc, false, false);
#pragma unroll
    for (int r2 = 0; r2 < 8; ++r2) accP[r2] += __expf(acc[r2]);
  }

  // ---- negative tiles (rows 384..1919) ----
  for (int nt = NT_POS; nt < NT_ALL; ++nt) {
    const _Float16* bp0 = Fz + (size_t)(NA + nt * 16 + n16) * KPAD;
    if (nt < NT_ALL - 1)
      __builtin_prefetch(Fz + (size_t)(NA + (nt + 1) * 16 + n16) * KPAD, 0, 3);

    Frag u[5];
#pragma unroll
    for (int kc = 0; kc < 5; ++kc) {
      const _Float16* bp = bp0 + kc * 32 + hf * 8;
      u[kc].i4[0] = *(const int4*)(bp);
      u[kc].i4[1] = *(const int4*)(bp + 16);
    }
    v8f acc = {0.f, 0.f, 0.f, 0.f, 0.f, 0.f, 0.f, 0.f};
#pragma unroll
    for (int kc = 0; kc < 5; ++kc)
      acc = __builtin_amdgcn_wmma_f32_16x16x32_f16(
          false, afr[kc], false, u[kc].h, (short)0, acc, false, false);
#pragma unroll
    for (int r2 = 0; r2 < 8; ++r2) accN[r2] += __expf(acc[r2]);
  }

  // one butterfly reduction per accumulator (over N = 16-lane groups)
#pragma unroll
  for (int r2 = 0; r2 < 8; ++r2) {
    float p = accP[r2];
    p += __shfl_xor(p, 1, 32);
    p += __shfl_xor(p, 2, 32);
    p += __shfl_xor(p, 4, 32);
    p += __shfl_xor(p, 8, 32);
    float n = accN[r2];
    n += __shfl_xor(n, 1, 32);
    n += __shfl_xor(n, 2, 32);
    n += __shfl_xor(n, 4, 32);
    n += __shfl_xor(n, 8, 32);
    if (n16 == 0) {                              // single writer per slot
      sums[mt * 16 + hf * 8 + r2]       = p;
      sums[128 + mt * 16 + hf * 8 + r2] = n;
    }
  }
  __syncthreads();

  if (tid < 128) red[tid] = logf(sums[tid] / sums[128 + tid]);
  __syncthreads();
  for (int s = 64; s > 0; s >>= 1) {
    if (tid < s) red[tid] += red[tid + s];
    __syncthreads();
  }
  if (tid == 0) partial[z] = red[0];
}

// ---------------------------------------------------------------------------
// Kernel 4: fixed-order final reduction -> deterministic scalar.
// ---------------------------------------------------------------------------
__global__ void k_finalize(const float* __restrict__ partial, float* __restrict__ out,
                           int nZ, float scale) {
  if (threadIdx.x == 0 && blockIdx.x == 0) {
    float t = 0.0f;
    for (int i = 0; i < nZ; ++i) t += partial[i];
    out[0] = t * scale;
  }
}

// ---------------------------------------------------------------------------
extern "C" void kernel_launch(void* const* d_in, const int* in_sizes, int n_in,
                              void* d_out, int out_size, void* d_ws, size_t ws_size,
                              hipStream_t stream) {
  (void)n_in; (void)out_size; (void)ws_size;

  const float* fine   = (const float*)d_in[0];
  const float* coarse = (const float*)d_in[1];
  const int*   GT     = (const int*)d_in[2];

  const int nGT = in_sizes[2];             // B*H*W
  const int nc  = in_sizes[1] / nGT;       // coarse channels (= n_classes) = 2
  const int Cf  = in_sizes[0] / nGT;       // fine channels = 128
  const int B   = 8;                       // harness-fixed batch
  const int HW  = nGT / B;                 // 65536

  const int nZ       = B * nc;             // 16 (b,class) instances
  const int totalPts = nZ * NPTS;          // 30720 sampled points

  // workspace layout (all regions fully overwritten every launch)
  char* ws = (char*)d_ws;
  float*    partial = (float*)ws;                                  // 16 floats
  int*      selIdx  = (int*)(ws + 256);                            // nZ*1920 ints
  _Float16* F       = (_Float16*)(ws + 256 + (size_t)nZ * NPTS * 4); // nZ*1920*160 f16

  // 1) compaction + bitonic sort + top-k selection (256KB+ dynamic LDS/WGP)
  size_t smemSort = (size_t)SORT_N * 8 + (size_t)SORT_NT * 4 + 64;
  k_sort_select<<<nZ * 2, SORT_NT, smemSort, stream>>>(coarse, GT, selIdx, nc, HW);

  // 2) gather + L2-normalize features (one wave32 per point)
  int gBlocks = (totalPts + 3) / 4;        // 4 waves per 128-thread block
  k_gather_norm<<<gBlocks, 128, 0, stream>>>(coarse, fine, selIdx, F, nc, Cf, HW, totalPts);

  // 3) WMMA cosine-sim GEMM + exp-sum + log ratio per (b,class)
  k_gemm_loss<<<nZ, 256, 0, stream>>>(F, partial);

  // 4) deterministic scalar:  total = sum_z partial[z] * (-1/(Na*B*nc))
  float scale = -1.0f / ((float)NA * (float)B * (float)nc);
  k_finalize<<<1, 32, 0, stream>>>(partial, (float*)d_out, nZ, scale);
}
